// NonLocalBlock_63290638074190
// MI455X (gfx1250) — matrix operand order repaired and checked by reading, not compiled, not run on hardware
//
#include <hip/hip_runtime.h>

// ---------------------------------------------------------------------------
// NonLocalBlock for MI455X (gfx1250): flash-style fused attention with
// v_wmma_f32_16x16x32_f16. Memory-bound on attention/H (256 MB @ 23.3 TB/s).
// v2: row sums computed by an extra P*ones WMMA (no shuffle reductions),
//     wave-uniform lazy-max rescale, i-range split across wave teams for 2x
//     occupancy with an LDS flash-merge.
// ---------------------------------------------------------------------------

typedef _Float16 v8h  __attribute__((ext_vector_type(8)));
typedef _Float16 v16h __attribute__((ext_vector_type(16)));
typedef float    v8f  __attribute__((ext_vector_type(8)));

constexpr int   BSZ = 2;
constexpr int   C   = 128;
constexpr int   NPT = 4096;
constexpr int   HID = 64;
constexpr float EPSV = 1e-5f;
constexpr float INV_SQRT_C = 0.08838834764831845f;   // 1/sqrt(128)

// ---------------------------------------------------------------------------
// Kernel 1: Q/K/V 1x1 conv projections.
//   Qt [b][n][c] f16 (pre-scaled by 1/sqrt(C))  -> WMMA A fragments
//   Kt [b][n][c] f16                             -> WMMA B fragments (S gemm)
//   Vh [b][c][n] f16                             -> WMMA B fragments (PV gemm)
// ---------------------------------------------------------------------------
__global__ __launch_bounds__(256) void proj_qkv(
    const float* __restrict__ feat,
    const float* __restrict__ Wq, const float* __restrict__ bq,
    const float* __restrict__ Wk, const float* __restrict__ bk,
    const float* __restrict__ Wv, const float* __restrict__ bv,
    _Float16* __restrict__ Qt, _Float16* __restrict__ Kt,
    _Float16* __restrict__ Vh)
{
  __shared__ float fcol[32][C + 4];           // padded: no bank conflicts
  const int b  = blockIdx.y;
  const int n0 = blockIdx.x * 32;
  const int t  = threadIdx.x;
  const float* fb = feat + (size_t)b * C * NPT;

  for (int idx = t; idx < 32 * C; idx += 256) {
    int c = idx >> 5, j = idx & 31;
    fcol[j][c] = fb[(size_t)c * NPT + n0 + j];
  }
  __syncthreads();

  const int co = t & 127;        // output channel
  const int j0 = t >> 7;         // 0..1
  const float* wq = Wq + co * C;
  const float* wk = Wk + co * C;
  const float* wv = Wv + co * C;

  for (int j = j0; j < 32; j += 2) {
    float aq = bq[co], ak = bk[co], av = bv[co];
    #pragma unroll 8
    for (int ci = 0; ci < C; ++ci) {
      float f = fcol[j][ci];
      aq += wq[ci] * f;
      ak += wk[ci] * f;
      av += wv[ci] * f;
    }
    const int n = n0 + j;
    Qt[((size_t)b * NPT + n) * C + co] = (_Float16)(aq * INV_SQRT_C);
    Kt[((size_t)b * NPT + n) * C + co] = (_Float16)ak;
    Vh[((size_t)b * C + co) * NPT + n] = (_Float16)av;
  }
}

// ---------------------------------------------------------------------------
// Kernel 2: fused  S = QtK / masked-scale / online softmax / O += P*V^T.
// Block = 256 threads = 8 waves. Waves 0-3 and 4-7 own the same 64 output
// rows but scan disjoint halves of the i range (flash split-K); results are
// merged through LDS. Each wave: 16 rows, i-tiles of 32.
// Per tile per wave: 8 WMMA (S) + 8 WMMA (PV) + 1 WMMA (row sums).
// ---------------------------------------------------------------------------
__global__ __launch_bounds__(256) void attn_fused(
    const _Float16* __restrict__ Qt, const _Float16* __restrict__ Kt,
    const _Float16* __restrict__ Vh,
    const float* __restrict__ attention, const float* __restrict__ Hmask,
    float* __restrict__ msg)
{
  __shared__ __align__(16) _Float16 pbuf[8][16 * 32];  // per-wave P staging
  __shared__ float obuf[4][16 * 128];                  // upper-team O tiles
  __shared__ float sbuf[4][16];                        // upper-team row sums
  __shared__ float mbuf[4];                            // upper-team maxima

  const int b    = blockIdx.y;
  const int wave = threadIdx.x >> 5;
  const int lane = threadIdx.x & 31;
  const int hf   = lane >> 4;          // half-wave: 0 or 1
  const int l16  = lane & 15;
  const int koff8 = hf * 16;           // A-fragment K sub-offset
  const int wrow = wave & 3;           // row-tile slot within block
  const int team = wave >> 2;          // which half of the i range
  const int obase = blockIdx.x * 64 + wrow * 16;
  const int ibeg = team * (NPT / 2);
  const int iend = ibeg + (NPT / 2);

  const _Float16* Qtb = Qt + (size_t)b * NPT * C;
  const _Float16* Ktb = Kt + (size_t)b * NPT * C;
  const _Float16* Vhb = Vh + (size_t)b * C * NPT;
  const float*  attb  = attention + (size_t)b * NPT * NPT;
  const float*  hb    = Hmask     + (size_t)b * NPT * NPT;

  // Preload Q A-fragments: 16 rows x K=128 in 4 chunks of 32.
  // A layout (16x32 f16): lane -> M=lane%16; halves 0..7 = K koff8..+7,
  // halves 8..15 = K koff8+8..+15.
  v16h aq[4];
  #pragma unroll
  for (int k = 0; k < 4; ++k) {
    const _Float16* p = &Qtb[(size_t)(obase + l16) * C + k * 32 + koff8];
    v8h lo = *(const v8h*)p;
    v8h hi = *(const v8h*)(p + 8);
    aq[k] = __builtin_shufflevector(lo, hi,
              0,1,2,3,4,5,6,7,8,9,10,11,12,13,14,15);
  }

  // constant all-ones B fragment: P x ones -> per-row sums in every lane
  v16h ones;
  #pragma unroll
  for (int i = 0; i < 16; ++i) ones[i] = (_Float16)1.0f;

  const v8f vzero = {0.f,0.f,0.f,0.f,0.f,0.f,0.f,0.f};
  v8f oacc[8];                      // O: 16 rows x 128 cols (8 c-tiles)
  v8f osum = vzero;                 // row-sum accumulator (same rescaling)
  float rmax = -3.0e38f;            // wave-uniform running max
  #pragma unroll
  for (int ct = 0; ct < 8; ++ct) oacc[ct] = vzero;

  for (int i0 = ibeg; i0 < iend; i0 += 32) {
    // ---- S tile (16 x 32) = Qt(16xK) * Kt^T ----
    v8f s0 = vzero, s1 = vzero;
    #pragma unroll
    for (int k = 0; k < 4; ++k) {
      // B layout (32x16 f16): lane -> N=lane%16, 16 contiguous K per half.
      const v16h bk0 = *(const v16h*)
          &Ktb[(size_t)(i0 + l16) * C + k * 32 + hf * 16];
      const v16h bk1 = *(const v16h*)
          &Ktb[(size_t)(i0 + 16 + l16) * C + k * 32 + hf * 16];
      s0 = __builtin_amdgcn_wmma_f32_16x16x32_f16(
              false, aq[k], false, bk0, (short)0, s0, false, false);
      s1 = __builtin_amdgcn_wmma_f32_16x16x32_f16(
              false, aq[k], false, bk1, (short)0, s1, false, false);
    }

    const bool more = (i0 + 32) < iend;   // uniform

    // ---- elementwise attention*S + mask; tile-wide max ----
    // D layout: reg r, lane -> row = r + 8*hf, col = l16 (s0) / l16+16 (s1)
    float tmax = -3.0e38f;
    #pragma unroll
    for (int r = 0; r < 8; ++r) {
      const int orow = obase + r + hf * 8;
      const size_t ro = (size_t)orow * NPT + i0 + l16;
      const float a0 = attb[ro], a1 = attb[ro + 16];
      const float h0 = hb[ro],   h1 = hb[ro + 16];
      if (more) {                       // stream-ahead prefetch (uniform)
        __builtin_prefetch(&attb[ro + 32]);
        __builtin_prefetch(&hb[ro + 32]);
      }
      const float v0 = fmaf(a0, s0[r], (h0 < 0.5f ? -1.0e9f : 0.0f));
      const float v1 = fmaf(a1, s1[r], (h1 < 0.5f ? -1.0e9f : 0.0f));
      s0[r] = v0;
      s1[r] = v1;
      tmax = fmaxf(tmax, fmaxf(v0, v1));
    }
    #pragma unroll
    for (int off = 1; off < 32; off <<= 1)
      tmax = fmaxf(tmax, __shfl_xor(tmax, off, 32));

    // lazy rescale: only when the (wave-uniform) max actually grows
    if (tmax > rmax) {
      const float al = __expf(rmax - tmax);
      rmax = tmax;
      osum = osum * al;
      #pragma unroll
      for (int ct = 0; ct < 8; ++ct) oacc[ct] = oacc[ct] * al;
    }

    // ---- P = exp(v - rmax), staged to LDS as f16 ----
    #pragma unroll
    for (int r = 0; r < 8; ++r) {
      const float p0 = __expf(s0[r] - rmax);
      const float p1 = __expf(s1[r] - rmax);
      _Float16* pb = &pbuf[wave][(r + hf * 8) * 32 + l16];
      pb[0]  = (_Float16)p0;
      pb[16] = (_Float16)p1;
    }

    // ---- re-read P as an A fragment (C/D -> A relayout via LDS) ----
    const _Float16* pr = &pbuf[wave][l16 * 32 + koff8];
    const v8h plo = *(const v8h*)pr;
    const v8h phi = *(const v8h*)(pr + 8);
    const v16h pa = __builtin_shufflevector(plo, phi,
              0,1,2,3,4,5,6,7,8,9,10,11,12,13,14,15);

    // ---- O(16x128) += P(16x32) * V^T(32x128); sums += P * ones ----
    #pragma unroll
    for (int ct = 0; ct < 8; ++ct) {
      const v16h bv = *(const v16h*)
          &Vhb[(size_t)(ct * 16 + l16) * NPT + i0 + hf * 16];
      oacc[ct] = __builtin_amdgcn_wmma_f32_16x16x32_f16(
              false, pa, false, bv, (short)0, oacc[ct], false, false);
    }
    osum = __builtin_amdgcn_wmma_f32_16x16x32_f16(
              false, pa, false, ones, (short)0, osum, false, false);
  }

  // ---- flash merge of the two i-teams through LDS ----
  if (team == 1) {
    #pragma unroll
    for (int r = 0; r < 8; ++r) {
      const int mrow = r + hf * 8;
      #pragma unroll
      for (int ct = 0; ct < 8; ++ct)
        obuf[wrow][mrow * 128 + ct * 16 + l16] = oacc[ct][r];
      if (l16 == 0) sbuf[wrow][mrow] = osum[r];
    }
    if (lane == 0) mbuf[wrow] = rmax;
  }
  __syncthreads();
  if (team == 0) {
    const float rhi = mbuf[wrow];
    const float m12 = fmaxf(rmax, rhi);
    const float alo = __expf(rmax - m12);
    const float ahi = __expf(rhi  - m12);
    float* mb = msg + (size_t)b * NPT * C;
    #pragma unroll
    for (int r = 0; r < 8; ++r) {
      const int mrow = r + hf * 8;
      const float ssum = osum[r] * alo + sbuf[wrow][mrow] * ahi;
      const float inv  = 1.0f / ssum;
      const int orow = obase + mrow;
      #pragma unroll
      for (int ct = 0; ct < 8; ++ct) {
        const float ohi = obuf[wrow][mrow * 128 + ct * 16 + l16];
        mb[(size_t)orow * C + ct * 16 + l16] =
            (oacc[ct][r] * alo + ohi * ahi) * inv;
      }
    }
  }
}

// ---------------------------------------------------------------------------
// Kernel 3: fc_message (conv-bn-relu x2 -> conv) + residual.
// One block handles 16 positions; msg rows are contiguous [n][c].
// ---------------------------------------------------------------------------
__global__ __launch_bounds__(256) void fc_residual(
    const float* __restrict__ msg,    // [b][n][c]
    const float* __restrict__ feat,
    const float* __restrict__ W1, const float* __restrict__ b1,
    const float* __restrict__ g1, const float* __restrict__ be1,
    const float* __restrict__ W2, const float* __restrict__ b2,
    const float* __restrict__ g2, const float* __restrict__ be2,
    const float* __restrict__ W3, const float* __restrict__ b3,
    float* __restrict__ out)
{
  __shared__ float sm[16][C + 1];
  __shared__ float m1[16][HID + 1];
  __shared__ float m2[16][HID + 1];
  const int b  = blockIdx.y;
  const int o0 = blockIdx.x * 16;
  const int t  = threadIdx.x;
  const float bnrs = rsqrtf(1.0f + EPSV);

  const float* mbase = msg + (size_t)b * NPT * C;
  for (int idx = t; idx < 16 * C; idx += 256) {
    int j = idx >> 7, c = idx & 127;
    sm[j][c] = mbase[(size_t)(o0 + j) * C + c];
  }
  __syncthreads();

  {
    const int h = t & 63, jb = t >> 6;
    const float* w = W1 + h * C;
    const float sc = g1[h] * bnrs, bb = b1[h], bt = be1[h];
    for (int j = jb; j < 16; j += 4) {
      float acc = bb;
      #pragma unroll 8
      for (int c = 0; c < C; ++c) acc += w[c] * sm[j][c];
      m1[j][h] = fmaxf(acc * sc + bt, 0.0f);
    }
  }
  __syncthreads();
  {
    const int h = t & 63, jb = t >> 6;
    const float* w = W2 + h * HID;
    const float sc = g2[h] * bnrs, bb = b2[h], bt = be2[h];
    for (int j = jb; j < 16; j += 4) {
      float acc = bb;
      #pragma unroll 8
      for (int c = 0; c < HID; ++c) acc += w[c] * m1[j][c];
      m2[j][h] = fmaxf(acc * sc + bt, 0.0f);
    }
  }
  __syncthreads();
  {
    const int j = t & 15, cb = t >> 4;
    const float* fb = feat + (size_t)b * C * NPT;
    float*       ob = out  + (size_t)b * C * NPT;
    for (int c = cb; c < C; c += 16) {
      float acc = b3[c];
      const float* w = W3 + c * HID;
      #pragma unroll 8
      for (int h = 0; h < HID; ++h) acc += w[h] * m2[j][h];
      const size_t idx = (size_t)c * NPT + o0 + j;
      ob[idx] = fb[idx] + acc;
    }
  }
}

// ---------------------------------------------------------------------------
extern "C" void kernel_launch(void* const* d_in, const int* in_sizes, int n_in,
                              void* d_out, int out_size, void* d_ws,
                              size_t ws_size, hipStream_t stream)
{
  (void)in_sizes; (void)n_in; (void)out_size; (void)ws_size;
  const float* feat      = (const float*)d_in[0];
  const float* attention = (const float*)d_in[1];
  const float* Wq = (const float*)d_in[2];
  const float* bq = (const float*)d_in[3];
  const float* Wk = (const float*)d_in[4];
  const float* bk = (const float*)d_in[5];
  const float* Wv = (const float*)d_in[6];
  const float* bv = (const float*)d_in[7];
  const float* W1 = (const float*)d_in[8];
  const float* b1 = (const float*)d_in[9];
  const float* g1 = (const float*)d_in[10];
  const float* be1 = (const float*)d_in[11];
  const float* W2 = (const float*)d_in[12];
  const float* b2 = (const float*)d_in[13];
  const float* g2 = (const float*)d_in[14];
  const float* be2 = (const float*)d_in[15];
  const float* W3 = (const float*)d_in[16];
  const float* b3 = (const float*)d_in[17];
  const float* Hm = (const float*)d_in[18];
  float* out = (float*)d_out;

  // workspace carve-out (~10 MB total)
  char* ws = (char*)d_ws;
  _Float16* Qt = (_Float16*)ws;  ws += (size_t)BSZ * NPT * C * sizeof(_Float16);
  _Float16* Kt = (_Float16*)ws;  ws += (size_t)BSZ * NPT * C * sizeof(_Float16);
  _Float16* Vh = (_Float16*)ws;  ws += (size_t)BSZ * NPT * C * sizeof(_Float16);
  float*    msg = (float*)ws;    // BSZ*NPT*C f32

  proj_qkv<<<dim3(NPT / 32, BSZ), 256, 0, stream>>>(
      feat, Wq, bq, Wk, bk, Wv, bv, Qt, Kt, Vh);
  attn_fused<<<dim3(NPT / 64, BSZ), 256, 0, stream>>>(
      Qt, Kt, Vh, attention, Hm, msg);
  fc_residual<<<dim3(NPT / 16, BSZ), 256, 0, stream>>>(
      msg, feat, W1, b1, g1, be1, W2, b2, g2, be2, W3, b3, out);
}
